// COMET_68813966017138
// MI455X (gfx1250) — matrix operand order repaired and checked by reading, not compiled, not run on hardware
//
#include <hip/hip_runtime.h>
#include <hip/hip_bf16.h>
#include <cstdint>

typedef __attribute__((ext_vector_type(16))) _Float16 v16h;
typedef __attribute__((ext_vector_type(8)))  _Float16 v8h;
typedef __attribute__((ext_vector_type(4)))  _Float16 v4h;
typedef __attribute__((ext_vector_type(8)))  float    v8f;
typedef _Float16 h16;

#define B_    32
#define N_    128
#define T_    192
#define L_    23
#define D_    512
#define H_    8
#define HD_   64
#define FF_   2048
#define K_    16
#define PRED_ 12
#define ROWS_ (B_*N_*L_)   /* 94208 token rows */
#define SEQS_ (B_*N_)      /* 4096 sequences   */

__device__ __forceinline__ float gelu_f(float x) {
  float x3 = x * x * x;
  return 0.5f * x * (1.0f + tanhf(0.7978845608028654f * (x + 0.044715f * x3)));
}

__device__ __forceinline__ v16h cat8(v8h lo, v8h hi) {
  return __builtin_shufflevector(lo, hi, 0, 1, 2, 3, 4, 5, 6, 7, 8, 9, 10, 11, 12, 13, 14, 15);
}

// ---------------------------------------------------------------------------
// Tiled WMMA GEMM: C[M,N] = epilogue(A[M,K] @ B[K,N])
// Block tile 64x64, BK=32, 256 threads = 8 waves (2 M-waves x 4 N-waves).
// AF16: A stored as f16 (one v8h global load/thread/k-step, no conversion)
// OF16: C stored as f16.  B (weights) always f32, L2-resident.
// B tile stored transposed in LDS so all fragments are ds_load_b128 runs.
// ---------------------------------------------------------------------------
template<bool AF16, bool OF16>
__global__ void __launch_bounds__(256)
gemm_wmma_kernel(const void* __restrict__ Ap, const float* __restrict__ Bm,
                 const float* __restrict__ bias, const float* __restrict__ resid,
                 void* __restrict__ Cp, int M, int N, int Kd, int act)
{
  const float* A32 = (const float*)Ap;
  const h16*   A16 = (const h16*)Ap;
  float* C32 = (float*)Cp;
  h16*   C16 = (h16*)Cp;

  __shared__ __align__(16) h16 As[64][40];   // [row][k], pitch 80 B
  __shared__ __align__(16) h16 Bs[64][40];   // TRANSPOSED: [col][k], pitch 80 B
  int tid = threadIdx.x;
  int m0 = blockIdx.y * 64, n0 = blockIdx.x * 64;
  bool fullM = (m0 + 64 <= M);
  bool fullN = (n0 + 64 <= N);

  int ln = tid & 31, wv = tid >> 5;
  int wm = wv >> 2, wn = wv & 3;      // wm in 0..1 (32 rows), wn in 0..3 (16 cols)
  int half = ln >> 4, lr = ln & 15;

  v8f acc0 = {};
  v8f acc1 = {};

  for (int k0 = 0; k0 < Kd; k0 += 32) {
    bool fullK = (k0 + 32 <= Kd);
    // ---- stage A tile (64 rows x 32 k)
    if (fullM && fullK) {
      if (AF16) {
        int r = tid >> 2, k8 = (tid & 3) * 8;               // 256 v8h
        const h16* src = A16 + (size_t)(m0 + r) * Kd + k0 + k8;
        *(v8h*)&As[r][k8] = *(const v8h*)src;
        __builtin_prefetch(src + 32, 0, 1);                 // next k-step
      } else {
#pragma unroll
        for (int i = 0; i < 2; ++i) {
          int f = tid + i * 256;                            // 512 float4
          int r = f >> 3, k4 = (f & 7) * 4;
          const float* src = A32 + (size_t)(m0 + r) * Kd + k0 + k4;
          const float4 v = *(const float4*)src;
          v4h h4 = { (h16)v.x, (h16)v.y, (h16)v.z, (h16)v.w };
          *(v4h*)&As[r][k4] = h4;
          __builtin_prefetch(src + 32, 0, 1);
        }
      }
    } else {
      for (int idx = tid; idx < 64 * 32; idx += 256) {
        int r = idx >> 5, k = idx & 31;
        int gm = m0 + r, gk = k0 + k;
        bool in = (gm < M && gk < Kd);
        if (AF16) As[r][k] = in ? A16[(size_t)gm * Kd + gk] : (h16)0;
        else      As[r][k] = (h16)(in ? A32[(size_t)gm * Kd + gk] : 0.0f);
      }
    }
    // ---- stage B tile (32 k x 64 cols), transposed into Bs[col][k]
    if (fullN && fullK) {
#pragma unroll
      for (int i = 0; i < 2; ++i) {
        int f = tid + i * 256;               // 512 float4
        int r = f >> 4, c4 = (f & 15) * 4;   // r = k row, c4 = col
        const float* src = Bm + (size_t)(k0 + r) * N + n0 + c4;
        const float4 v = *(const float4*)src;
        Bs[c4 + 0][r] = (h16)v.x;
        Bs[c4 + 1][r] = (h16)v.y;
        Bs[c4 + 2][r] = (h16)v.z;
        Bs[c4 + 3][r] = (h16)v.w;
        __builtin_prefetch(src + 32 * (size_t)N, 0, 1);
      }
    } else {
      for (int idx = tid; idx < 32 * 64; idx += 256) {
        int r = idx >> 6, c = idx & 63;
        int gk = k0 + r, gn = n0 + c;
        float v = (gk < Kd && gn < N) ? Bm[(size_t)gk * N + gn] : 0.0f;
        Bs[c][r] = (h16)v;
      }
    }
    __syncthreads();

    // ---- fragments (contiguous 16-byte LDS runs -> ds_load_b128)
    int rowA0 = wm * 32 + lr, rowA1 = rowA0 + 16;
    v16h a0 = cat8(*(const v8h*)&As[rowA0][half * 8],
                   *(const v8h*)&As[rowA0][16 + half * 8]);
    v16h a1 = cat8(*(const v8h*)&As[rowA1][half * 8],
                   *(const v8h*)&As[rowA1][16 + half * 8]);
    int colB = wn * 16 + lr;
    v16h b0 = cat8(*(const v8h*)&Bs[colB][half * 8],
                   *(const v8h*)&Bs[colB][16 + half * 8]);

    acc0 = __builtin_amdgcn_wmma_f32_16x16x32_f16(false, a0, false, b0, (short)0, acc0, false, false);
    acc1 = __builtin_amdgcn_wmma_f32_16x16x32_f16(false, a1, false, b0, (short)0, acc1, false, false);
    __syncthreads();
  }

  // Epilogue. D layout: VGPR r -> lanes 0-15 M=r, lanes 16-31 M=8+r.
  int nn = n0 + wn * 16 + lr;
#pragma unroll
  for (int r = 0; r < 8; ++r) {
    int mmA = m0 + wm * 32 + r + half * 8;
    int mmB = mmA + 16;
    float v0 = acc0[r], v1 = acc1[r];
    if (nn < N) {
      if (bias) { float bb = bias[nn]; v0 += bb; v1 += bb; }
      if (act)  { v0 = gelu_f(v0); v1 = gelu_f(v1); }
      if (mmA < M) {
        if (resid) v0 += resid[(size_t)mmA * N + nn];
        if (OF16) C16[(size_t)mmA * N + nn] = (h16)v0;
        else      C32[(size_t)mmA * N + nn] = v0;
      }
      if (mmB < M) {
        if (resid) v1 += resid[(size_t)mmB * N + nn];
        if (OF16) C16[(size_t)mmB * N + nn] = (h16)v1;
        else      C32[(size_t)mmB * N + nn] = v1;
      }
    }
  }
}

// ---------------------------------------------------------------------------
// Attention: one wave per (sequence, head). q/k/v already f16 in global.
// scores = q k^T / 8 (8 WMMAs), softmax in LDS, out = a v (8 WMMAs).
// ---------------------------------------------------------------------------
__global__ void __launch_bounds__(32)
attn_kernel(const h16* __restrict__ Qm, const h16* __restrict__ Km,
            const h16* __restrict__ Vm, h16* __restrict__ Om)
{
  __shared__ __align__(16) h16   qs[32][72];   // [l][d], pitch 144 B
  __shared__ __align__(16) h16   ks[32][72];   // [l][d]
  __shared__ __align__(16) h16   vsT[64][40];  // TRANSPOSED: [d][l], pitch 80 B
  __shared__ __align__(16) float sb[32][33];
  __shared__ __align__(16) h16   ab[32][40];   // probs, pitch 80 B

  int tid = threadIdx.x;
  int sh = blockIdx.x;
  int seq = sh >> 3, head = sh & 7;
  size_t base = ((size_t)seq * L_) * D_ + (size_t)head * HD_;

#pragma unroll
  for (int i = 0; i < 8; ++i) {
    int f = tid + i * 32;                 // 256 v8h per matrix
    int r = f >> 3, c8 = (f & 7) * 8;
    v8h q = {}, k = {}, v = {};
    if (r < L_) {
      size_t g = base + (size_t)r * D_ + c8;
      q = *(const v8h*)(Qm + g);
      k = *(const v8h*)(Km + g);
      v = *(const v8h*)(Vm + g);
    }
    *(v8h*)&qs[r][c8] = q;
    *(v8h*)&ks[r][c8] = k;
#pragma unroll
    for (int j = 0; j < 8; ++j) vsT[c8 + j][r] = v[j];
  }
  __syncthreads();

  int half = tid >> 4, lr = tid & 15;

  // ---- scores
  v8f sacc[2][2] = {};
#pragma unroll
  for (int kstep = 0; kstep < 2; ++kstep) {
    int k0 = kstep * 32;
    v16h a0 = cat8(*(const v8h*)&qs[lr][k0 + half * 8],
                   *(const v8h*)&qs[lr][k0 + 16 + half * 8]);
    v16h a1 = cat8(*(const v8h*)&qs[16 + lr][k0 + half * 8],
                   *(const v8h*)&qs[16 + lr][k0 + 16 + half * 8]);
    v16h b0 = cat8(*(const v8h*)&ks[lr][k0 + half * 8],
                   *(const v8h*)&ks[lr][k0 + 16 + half * 8]);
    v16h b1 = cat8(*(const v8h*)&ks[16 + lr][k0 + half * 8],
                   *(const v8h*)&ks[16 + lr][k0 + 16 + half * 8]);
    sacc[0][0] = __builtin_amdgcn_wmma_f32_16x16x32_f16(false, a0, false, b0, (short)0, sacc[0][0], false, false);
    sacc[0][1] = __builtin_amdgcn_wmma_f32_16x16x32_f16(false, a0, false, b1, (short)0, sacc[0][1], false, false);
    sacc[1][0] = __builtin_amdgcn_wmma_f32_16x16x32_f16(false, a1, false, b0, (short)0, sacc[1][0], false, false);
    sacc[1][1] = __builtin_amdgcn_wmma_f32_16x16x32_f16(false, a1, false, b1, (short)0, sacc[1][1], false, false);
  }
#pragma unroll
  for (int mi = 0; mi < 2; ++mi)
#pragma unroll
    for (int ni = 0; ni < 2; ++ni)
#pragma unroll
      for (int r = 0; r < 8; ++r)
        sb[mi * 16 + half * 8 + r][ni * 16 + lr] = sacc[mi][ni][r] * 0.125f;
  __syncthreads();

  // ---- softmax over valid 23 cols; lane = row
  if (tid < L_) {
    float mx = -1e30f;
    for (int n = 0; n < L_; ++n) mx = fmaxf(mx, sb[tid][n]);
    float sum = 0.f;
    for (int n = 0; n < L_; ++n) { float e = __expf(sb[tid][n] - mx); sum += e; sb[tid][n] = e; }
    float inv = 1.0f / sum;
    for (int n = 0; n < 32; ++n) ab[tid][n] = (h16)(n < L_ ? sb[tid][n] * inv : 0.0f);
  } else {
    for (int n = 0; n < 32; ++n) ab[tid][n] = (h16)0.0f;
  }
  __syncthreads();

  // ---- out = a @ v
  v8f oacc[2][4] = {};
  v16h a0 = cat8(*(const v8h*)&ab[lr][half * 8],
                 *(const v8h*)&ab[lr][16 + half * 8]);
  v16h a1 = cat8(*(const v8h*)&ab[16 + lr][half * 8],
                 *(const v8h*)&ab[16 + lr][16 + half * 8]);
#pragma unroll
  for (int ni = 0; ni < 4; ++ni) {
    v16h bv = cat8(*(const v8h*)&vsT[ni * 16 + lr][half * 8],
                   *(const v8h*)&vsT[ni * 16 + lr][16 + half * 8]);
    oacc[0][ni] = __builtin_amdgcn_wmma_f32_16x16x32_f16(false, a0, false, bv, (short)0, oacc[0][ni], false, false);
    oacc[1][ni] = __builtin_amdgcn_wmma_f32_16x16x32_f16(false, a1, false, bv, (short)0, oacc[1][ni], false, false);
  }
#pragma unroll
  for (int mi = 0; mi < 2; ++mi)
#pragma unroll
    for (int ni = 0; ni < 4; ++ni)
#pragma unroll
      for (int r = 0; r < 8; ++r) {
        int m = mi * 16 + half * 8 + r;
        if (m < L_) Om[base + (size_t)m * D_ + ni * 16 + lr] = (h16)oacc[mi][ni][r];
      }
}

// ---------------------------------------------------------------------------
// LayerNorm over D=512 (fp32 in, f16 out), one 256-thread block per row.
// ---------------------------------------------------------------------------
__global__ void __launch_bounds__(256)
ln_kernel(const float* __restrict__ x, const float* __restrict__ g,
          const float* __restrict__ bb, h16* __restrict__ y)
{
  __shared__ float red[256];
  __shared__ float s_mean, s_var;
  int row = blockIdx.x, t = threadIdx.x;
  const float* xr = x + (size_t)row * D_;
  float v1 = xr[t], v2 = xr[t + 256];
  red[t] = v1 + v2; __syncthreads();
  for (int s = 128; s > 0; s >>= 1) { if (t < s) red[t] += red[t + s]; __syncthreads(); }
  if (t == 0) s_mean = red[0] * (1.0f / D_);
  __syncthreads();
  float m = s_mean;
  float d1 = v1 - m, d2 = v2 - m;
  red[t] = d1 * d1 + d2 * d2; __syncthreads();
  for (int s = 128; s > 0; s >>= 1) { if (t < s) red[t] += red[t + s]; __syncthreads(); }
  if (t == 0) s_var = red[0] * (1.0f / D_);
  __syncthreads();
  float rstd = rsqrtf(s_var + 1e-5f);
  h16* yr = y + (size_t)row * D_;
  yr[t]       = (h16)(d1 * rstd * g[t]       + bb[t]);
  yr[t + 256] = (h16)(d2 * rstd * g[t + 256] + bb[t + 256]);
}

// ---------------------------------------------------------------------------
__global__ void sort_kernel(const int* __restrict__ obs, int* __restrict__ obs_idx,
                            int* __restrict__ inv_idx, int* __restrict__ n_obs)
{
  int b = blockIdx.x;
  if (threadIdx.x != 0) return;
  int cnt = 0;
  for (int n = 0; n < N_; ++n)
    if (obs[b * N_ + n] != 0) { obs_idx[b * N_ + cnt] = n; inv_idx[b * N_ + n] = cnt; cnt++; }
  n_obs[b] = cnt;
  int c2 = cnt;
  for (int n = 0; n < N_; ++n)
    if (obs[b * N_ + n] == 0) { obs_idx[b * N_ + c2] = n; inv_idx[b * N_ + n] = c2; c2++; }
}

__global__ void patch_kernel(const float* __restrict__ x_full, const int* __restrict__ obs,
                             float* __restrict__ P)
{
  int idx = blockIdx.x * 256 + threadIdx.x;
  if (idx >= ROWS_ * 16) return;
  int t = idx & 15;
  int row = idx >> 4;
  int l = row % L_;
  int bn = row / L_;
  float v = 0.f;
  if (obs[bn] != 0) v = x_full[(size_t)bn * T_ + l * 8 + t];
  P[idx] = v;
}

__global__ void gather_tok_kernel(const float* __restrict__ h, const int* __restrict__ obs_idx,
                                  const float* __restrict__ ve, float* __restrict__ tok)
{
  size_t idx = (size_t)blockIdx.x * 256 + threadIdx.x;
  if (idx >= (size_t)ROWS_ * D_) return;
  int d = (int)(idx & (D_ - 1));
  size_t row = idx >> 9;
  int l = (int)(row % L_);
  size_t bi = row / L_;
  int b = (int)(bi / N_), i = (int)(bi % N_);
  int src = obs_idx[b * N_ + i];
  tok[idx] = h[(((size_t)(b * N_ + src)) * L_ + l) * D_ + d] + ve[(size_t)src * D_ + d];
}

__global__ void pool_kernel(const float* __restrict__ tok, const int* __restrict__ n_obs,
                            float* __restrict__ Qsub)
{
  int b = blockIdx.x;
  int d = blockIdx.y * 256 + threadIdx.x;
  int no = n_obs[b];
  float acc = 0.f;
  for (int i = 0; i < no; ++i) {
    const float* p = tok + (((size_t)(b * N_ + i)) * L_) * D_ + d;
    for (int l = 0; l < L_; ++l) acc += p[(size_t)l * D_];
  }
  Qsub[b * D_ + d] = acc / (float)(no * L_);
}

__global__ void __launch_bounds__(256)
vq_kernel(const float* __restrict__ Qsub, const float* __restrict__ Cb,
          float* __restrict__ wsub, float* __restrict__ code)
{
  __shared__ float red[256], redc[256], dd[16], wv[16];
  __shared__ float qq_s;
  int b = blockIdx.x, t = threadIdx.x;
  float q1 = Qsub[b * D_ + t], q2 = Qsub[b * D_ + 256 + t];
  red[t] = q1 * q1 + q2 * q2; __syncthreads();
  for (int s = 128; s > 0; s >>= 1) { if (t < s) red[t] += red[t + s]; __syncthreads(); }
  if (t == 0) qq_s = red[0];
  __syncthreads();
  float qq = qq_s;
  for (int k = 0; k < K_; ++k) {
    float c1 = Cb[k * D_ + t], c2 = Cb[k * D_ + 256 + t];
    red[t] = q1 * c1 + q2 * c2;
    redc[t] = c1 * c1 + c2 * c2;
    __syncthreads();
    for (int s = 128; s > 0; s >>= 1) {
      if (t < s) { red[t] += red[t + s]; redc[t] += redc[t + s]; }
      __syncthreads();
    }
    if (t == 0) dd[k] = qq - 2.0f * red[0] + redc[0];
    __syncthreads();
  }
  if (t == 0) {
    float mx = -1e30f;
    for (int k = 0; k < K_; ++k) { float lg = -dd[k] * 2.0f; if (lg > mx) mx = lg; }  // TAU=0.5
    float sum = 0.f;
    for (int k = 0; k < K_; ++k) { float e = __expf(-dd[k] * 2.0f - mx); wv[k] = e; sum += e; }
    float inv = 1.0f / sum;
    for (int k = 0; k < K_; ++k) { wv[k] *= inv; wsub[b * K_ + k] = wv[k]; }
  }
  __syncthreads();
  for (int d = t; d < D_; d += 256) {
    float s = 0.f;
    for (int k = 0; k < K_; ++k) s += wv[k] * Cb[k * D_ + d];
    code[b * D_ + d] = s;
  }
}

__global__ void build_min_kernel(const float* __restrict__ ve, const float* __restrict__ code,
                                 const float* __restrict__ Qsub, float* __restrict__ Min)
{
  int idx = blockIdx.x * 256 + threadIdx.x;
  if (idx >= SEQS_ * D_) return;
  int d = idx & (D_ - 1);
  int bn = idx >> 9;
  int b = bn / N_, n = bn % N_;
  Min[idx] = ve[n * D_ + d] + code[b * D_ + d] + Qsub[b * D_ + d];
}

__global__ void build_e_kernel(const float* __restrict__ tok, const float* __restrict__ miss,
                               const int* __restrict__ obs, const int* __restrict__ inv,
                               h16* __restrict__ E)
{
  size_t idx = (size_t)blockIdx.x * 256 + threadIdx.x;
  if (idx >= (size_t)ROWS_ * D_) return;
  int d = (int)(idx & (D_ - 1));
  size_t row = idx >> 9;
  int l = (int)(row % L_);
  size_t bn = row / L_;
  int b = (int)(bn / N_), n = (int)(bn % N_);
  float v;
  if (obs[b * N_ + n] != 0) {
    int i = inv[b * N_ + n];
    v = tok[(((size_t)(b * N_ + i)) * L_ + l) * D_ + d];
  } else {
    v = miss[bn * D_ + d];
  }
  E[idx] = (h16)v;
}

// ---------------------------------------------------------------------------
template<bool AF16, bool OF16>
static void launch_gemm(const void* A, const float* Bm, const float* bias, const float* resid,
                        void* C, int M, int N, int K, int act, hipStream_t s)
{
  dim3 grid((N + 63) / 64, (M + 63) / 64);
  gemm_wmma_kernel<AF16, OF16><<<grid, dim3(256), 0, s>>>(A, Bm, bias, resid, C, M, N, K, act);
}

extern "C" void kernel_launch(void* const* d_in, const int* in_sizes, int n_in,
                              void* d_out, int out_size, void* d_ws, size_t ws_size,
                              hipStream_t stream)
{
  (void)in_sizes; (void)n_in; (void)out_size; (void)ws_size;
  // top-level dict insertion order; params flattened as jax pytree (alphabetical)
  const float* x_full   = (const float*)d_in[0];
  const int*   obs_mask = (const int*)  d_in[1];
  const float* Cb       = (const float*)d_in[2];
  const float* W_dec    = (const float*)d_in[3];
  const float* W_h1     = (const float*)d_in[4];
  const float* W_h2     = (const float*)d_in[5];
  const float* W_patch  = (const float*)d_in[6];
  const float* W_tp1    = (const float*)d_in[7];
  const float* W_tp2    = (const float*)d_in[8];
  const float* b_dec    = (const float*)d_in[9];
  const float* b_h1     = (const float*)d_in[10];
  const float* b_h2     = (const float*)d_in[11];
  const float* b_patch  = (const float*)d_in[12];
  const float* b_tp1    = (const float*)d_in[13];
  const float* b_tp2    = (const float*)d_in[14];
  const float* var_embed = (const float*)d_in[39];

  size_t big = (size_t)ROWS_ * D_;
  float* ws32 = (float*)d_ws;
  float* tok  = ws32;                             // f32 residual stream
  float* bufH = tok + big;                        // f32 h / scratch
  float* Min  = bufH + big;                       // 4096 x 512 f32
  float* miss = Min  + (size_t)SEQS_ * D_;        // 4096 x 512 f32
  float* G    = miss + (size_t)SEQS_ * D_;        // 4096 x 512 f32
  float* code = G    + (size_t)SEQS_ * D_;        // 32 x 512 f32
  int* obs_idx = (int*)(code + B_ * D_);
  int* inv_idx = obs_idx + B_ * N_;
  int* n_obs   = inv_idx + B_ * N_;
  uintptr_t hp = ((uintptr_t)(n_obs + B_) + 255) & ~(uintptr_t)255;
  h16* hX = (h16*)hp;          // f16 activations, each ROWS_ x 512
  h16* hQ = hX + big;
  h16* hK = hQ + big;
  h16* hV = hK + big;
  h16* hT = hV + big;          // tp/FF intermediate (f16)
  float* P = (float*)hQ;       // patches (f32, 6 MB) before hQ is live

  float* y_hat = (float*)d_out;                   // 4096 x 12
  float* Qsub  = y_hat + SEQS_ * PRED_;           // 32 x 512
  float* wsub  = Qsub + B_ * D_;                  // 32 x 16

  auto cdiv = [](int a, int b) { return (a + b - 1) / b; };

  sort_kernel<<<B_, 32, 0, stream>>>(obs_mask, obs_idx, inv_idx, n_obs);
  patch_kernel<<<cdiv(ROWS_ * 16, 256), 256, 0, stream>>>(x_full, obs_mask, P);
  launch_gemm<false, false>(P, W_patch, b_patch, nullptr, bufH, ROWS_, D_, 16, 0, stream);      // h
  launch_gemm<false, true >(bufH, W_tp1, b_tp1, nullptr, hT, ROWS_, D_, D_, 1, stream);         // gelu
  launch_gemm<true , false>(hT, W_tp2, b_tp2, bufH, bufH, ROWS_, D_, D_, 0, stream);            // h += ...

  {
    long long tot = (long long)ROWS_ * D_;
    gather_tok_kernel<<<(int)((tot + 255) / 256), 256, 0, stream>>>(bufH, obs_idx, var_embed, tok);
  }

  for (int layer = 0; layer < 2; ++layer) {
    int base = 15 + layer * 12;
    const float* W_ff1 = (const float*)d_in[base + 0];
    const float* W_ff2 = (const float*)d_in[base + 1];
    const float* Wk    = (const float*)d_in[base + 2];
    const float* Wo    = (const float*)d_in[base + 3];
    const float* Wq    = (const float*)d_in[base + 4];
    const float* Wv    = (const float*)d_in[base + 5];
    const float* b_ff1 = (const float*)d_in[base + 6];
    const float* b_ff2 = (const float*)d_in[base + 7];
    const float* ln1_b = (const float*)d_in[base + 8];
    const float* ln1_g = (const float*)d_in[base + 9];
    const float* ln2_b = (const float*)d_in[base + 10];
    const float* ln2_g = (const float*)d_in[base + 11];

    ln_kernel<<<ROWS_, 256, 0, stream>>>(tok, ln1_g, ln1_b, hX);
    launch_gemm<true, true>(hX, Wq, nullptr, nullptr, hQ, ROWS_, D_, D_, 0, stream);
    launch_gemm<true, true>(hX, Wk, nullptr, nullptr, hK, ROWS_, D_, D_, 0, stream);
    launch_gemm<true, true>(hX, Wv, nullptr, nullptr, hV, ROWS_, D_, D_, 0, stream);
    attn_kernel<<<SEQS_ * H_, 32, 0, stream>>>(hQ, hK, hV, hX);          // O -> hX
    launch_gemm<true, false>(hX, Wo, nullptr, tok, tok, ROWS_, D_, D_, 0, stream);   // tok += O Wo

    ln_kernel<<<ROWS_, 256, 0, stream>>>(tok, ln2_g, ln2_b, hX);
    const int CH = 8192;
    for (int r0 = 0; r0 < ROWS_; r0 += CH) {
      int mr = (ROWS_ - r0 < CH) ? (ROWS_ - r0) : CH;
      launch_gemm<true, true >(hX + (size_t)r0 * D_, W_ff1, b_ff1, nullptr, hT, mr, FF_, D_, 1, stream);
      launch_gemm<true, false>(hT, W_ff2, b_ff2, tok + (size_t)r0 * D_, tok + (size_t)r0 * D_, mr, D_, FF_, 0, stream);
    }
  }

  pool_kernel<<<dim3(B_, D_ / 256), 256, 0, stream>>>(tok, n_obs, Qsub);
  vq_kernel<<<B_, 256, 0, stream>>>(Qsub, Cb, wsub, code);
  build_min_kernel<<<cdiv(SEQS_ * D_, 256), 256, 0, stream>>>(var_embed, code, Qsub, Min);
  launch_gemm<false, false>(Min, W_dec, b_dec, nullptr, miss, SEQS_, D_, D_, 1, stream);
  {
    long long tot = (long long)ROWS_ * D_;
    build_e_kernel<<<(int)((tot + 255) / 256), 256, 0, stream>>>(tok, miss, obs_mask, inv_idx, hK);
  }
  launch_gemm<true , false>(hK, W_h1, b_h1, nullptr, G, SEQS_, D_, L_ * D_, 1, stream);  // gelu(E W_h1 + b)
  launch_gemm<false, false>(G, W_h2, b_h2, nullptr, y_hat, SEQS_, PRED_, D_, 0, stream); // y_hat
}
